// SimpleMultiplePassGNNReachabilityNet_13305808683173
// MI455X (gfx1250) — matrix operand order repaired and checked by reading, not compiled, not run on hardware
//
#include <hip/hip_runtime.h>
#include <hip/hip_bf16.h>

// ---------------------------------------------------------------------------
// SimpleMultiplePassGNNReachabilityNet forward, CDNA5 (gfx1250, wave32).
// Edge MLP and vertex layer 1 run on v_wmma_f32_16x16x32_f16 (f16 in, f32 acc); node MLP is scalar.
// h is materialized as f16 (halves L2 gather bytes; gathers/atomics are the
// bound, not FLOPs). Scatter-mean via f32 global atomics into L2-resident ws.
// The layer1->layer2 C->A transpose goes through LDS in a K-PERMUTED packed
// layout (pairs (n, n+16) stored contiguously) so each lane issues 8 packed
// DS stores instead of 16 scalar ones; the layer-2 weight tiles are packed
// through the inverse permutation, which leaves the GEMM result unchanged.
// ---------------------------------------------------------------------------

typedef __attribute__((ext_vector_type(16))) _Float16 v16h;
typedef __attribute__((ext_vector_type(8)))  _Float16 v8h;
typedef __attribute__((ext_vector_type(2)))  _Float16 v2h;
typedef __attribute__((ext_vector_type(8)))  float    v8f;
typedef __attribute__((ext_vector_type(2)))  float    v2f;

#define LEAKY_SLOPE 0.01f
#define WAVES_PER_BLOCK 4

__device__ __forceinline__ v8f wmma_f16_16x16x32(v16h a, v16h b, v8f c) {
  // 8 args: (neg_a, A, neg_b, B, c_mod, C, reuse_a, reuse_b)
  return __builtin_amdgcn_wmma_f32_16x16x32_f16(false, a, false, b, (short)0, c,
                                                false, false);
}

__device__ __forceinline__ v8f splat8(float x) {
  v8f c;
#pragma unroll
  for (int j = 0; j < 8; ++j) c[j] = x;
  return c;
}

// Stored-position -> original-feature permutation for the packed LDS tiles:
// position p holds original feature (p>>1) + 16*(p&1)   (order 0,16,1,17,...)
__device__ __forceinline__ int kperm(int p) { return (p >> 1) + 16 * (p & 1); }

// Pack a 32(K) x 16(N) f16 B-operand tile from a row-major f32 weight matrix
// W[out][in] (ld = in-dim). B[k][n] = W[n0+n][k0+k]. wave32 B layout:
// lanes 0-15 = col n, K 0..7 & 16..23; lanes 16-31 = col n, K 8..15 & 24..31.
__device__ __forceinline__ v16h make_btile(const float* __restrict__ W, int ld,
                                           int n0, int k0, int lane) {
  const int col = n0 + (lane & 15);
  const int kb  = k0 + ((lane >> 4) << 3);  // +8 for hi half-wave
  const float* wp = W + (size_t)col * ld + kb;
  v16h b;
#pragma unroll
  for (int v = 0; v < 8; ++v) {
    b[2 * v]     = (_Float16)wp[2 * v + ((v >> 2) << 3)];
    b[2 * v + 1] = (_Float16)wp[2 * v + 1 + ((v >> 2) << 3)];
  }
  return b;
}

// Same, but the K axis is indexed through kperm() to match the packed
// (n, n+16)-interleaved activation tiles in LDS. K-dim is 32 (k0 == 0).
__device__ __forceinline__ v16h make_btile_perm(const float* __restrict__ W,
                                                int ld, int n0, int lane) {
  const int col = n0 + (lane & 15);
  const int kb  = (lane >> 4) << 3;
  const float* wp = W + (size_t)col * ld;
  v16h b;
#pragma unroll
  for (int v = 0; v < 8; ++v) {
    b[2 * v]     = (_Float16)wp[kperm(kb + 2 * v + ((v >> 2) << 3))];
    b[2 * v + 1] = (_Float16)wp[kperm(kb + 2 * v + 1 + ((v >> 2) << 3))];
  }
  return b;
}

// A-operand tile row fetch (16x32 f16 A layout). `row` = base of one 32-f16
// feature row for this lane's matrix row (lane&15). Lane l<16 holds K 0..7 &
// 16..23; lane>=16 holds K 8..15 & 24..31  ->  two 16B chunks at +0/+32B from
// a lane-dependent 8-element base.
__device__ __forceinline__ v16h load_arow(const _Float16* __restrict__ row,
                                          int lane) {
  const _Float16* p = row + ((lane >> 4) << 3);
  v8h lo = *(const v8h*)(p);
  v8h hi = *(const v8h*)(p + 16);
  v16h a;
#pragma unroll
  for (int v = 0; v < 8; ++v) { a[v] = lo[v]; a[8 + v] = hi[v]; }
  return a;
}

// ---------------------------------------------------------------------------
// Stage 1: per-node MLP 11 -> 32 (relu) -> 32 (leaky), store h as f16.
// ---------------------------------------------------------------------------
__global__ void k_node_mlp(const float* __restrict__ verts,
                           const float* __restrict__ Wx1,
                           const float* __restrict__ bx1,
                           const float* __restrict__ Wx2,
                           const float* __restrict__ bx2,
                           _Float16* __restrict__ h16, int BN) {
  __shared__ float sW1[32 * 11];
  __shared__ float sW2[32 * 32];
  __shared__ float sb1[32];
  __shared__ float sb2[32];
  const int tid = threadIdx.x;
  for (int i = tid; i < 32 * 11; i += blockDim.x) sW1[i] = Wx1[i];
  for (int i = tid; i < 32 * 32; i += blockDim.x) sW2[i] = Wx2[i];
  for (int i = tid; i < 32; i += blockDim.x) { sb1[i] = bx1[i]; sb2[i] = bx2[i]; }
  __syncthreads();

  const int idx = blockIdx.x * blockDim.x + tid;
  if (idx >= BN) return;
  float x[11];
  const float* vp = verts + (size_t)idx * 11;
#pragma unroll
  for (int i = 0; i < 11; ++i) x[i] = vp[i];
  float h1[32];
#pragma unroll
  for (int c = 0; c < 32; ++c) {
    float acc = sb1[c];
#pragma unroll
    for (int i = 0; i < 11; ++i) acc = fmaf(sW1[c * 11 + i], x[i], acc);
    h1[c] = fmaxf(acc, 0.0f);
  }
  _Float16* hp = h16 + (size_t)idx * 32;
#pragma unroll
  for (int o = 0; o < 32; ++o) {
    float acc = sb2[o];
#pragma unroll
    for (int c = 0; c < 32; ++c) acc = fmaf(sW2[o * 32 + c], h1[c], acc);
    acc = acc > 0.0f ? acc : LEAKY_SLOPE * acc;
    hp[o] = (_Float16)acc;
  }
}

// ---------------------------------------------------------------------------
// Degree counts (batch-independent): each edge adds 1 to both endpoints.
// ---------------------------------------------------------------------------
__global__ void k_count(const int* __restrict__ e0, const int* __restrict__ e1,
                        float* __restrict__ cnt, int E) {
  const int e = blockIdx.x * blockDim.x + threadIdx.x;
  if (e < E) {
    atomicAdd(&cnt[e1[e]], 1.0f);
    atomicAdd(&cnt[e0[e]], 1.0f);
  }
}

// ---------------------------------------------------------------------------
// Stage 2: edge MLP (64->32 relu ->32 leaky) on WMMA + symmetric scatter-add.
// One wave per 16-edge tile; batch loop inside reuses edge indices + weights.
// ---------------------------------------------------------------------------
__global__ void __launch_bounds__(32 * WAVES_PER_BLOCK)
k_edge_mlp(const _Float16* __restrict__ h16, const int* __restrict__ e0,
           const int* __restrict__ e1, const float* __restrict__ We1,
           const float* __restrict__ be1, const float* __restrict__ We2,
           const float* __restrict__ be2, float* __restrict__ sums, int Nn,
           int Bb, int ntiles) {
  __shared__ __align__(16) _Float16 lds[WAVES_PER_BLOCK * 16 * 32];
  const int lane = threadIdx.x & 31;
  const int wib  = threadIdx.x >> 5;
  _Float16* myt  = &lds[wib * 512];
  const int gw = blockIdx.x * WAVES_PER_BLOCK + wib;
  const int nw = gridDim.x * WAVES_PER_BLOCK;

  // Weight tiles packed once per wave (L1/L2 resident).
  v16h B1[2][2];  // [k-tile 0..1][n-half 0..1] for We1 (32 x 64)
  v16h B2[2];     // [n-half] for We2 (32 x 32), K-permuted to packed layout
#pragma unroll
  for (int nt = 0; nt < 2; ++nt) {
#pragma unroll
    for (int kt = 0; kt < 2; ++kt) B1[kt][nt] = make_btile(We1, 64, nt * 16, kt * 32, lane);
    B2[nt] = make_btile_perm(We2, 32, nt * 16, lane);
  }
  const float b1lo = be1[lane & 15], b1hi = be1[16 + (lane & 15)];
  const float b2lo = be2[lane & 15], b2hi = be2[16 + (lane & 15)];

  for (int t = gw; t < ntiles; t += nw) {
    const int ebase = t * 16;
    const int arow  = ebase + (lane & 15);  // this lane's A-row edge
    const int nsrc  = e0[arow];
    const int ndst  = e1[arow];

    // Hoist per-lane scatter endpoints (batch-invariant): row = j + 8*(hi).
    int srow[8], drow[8];
#pragma unroll
    for (int j = 0; j < 8; ++j) {
      const int e = ebase + j + ((lane >> 4) << 3);
      srow[j] = e0[e];
      drow[j] = e1[e];
    }

    for (int b = 0; b < Bb; ++b) {
      const _Float16* hb = h16 + (size_t)b * Nn * 32;
      // pair = [h[src] | h[dst]] : K 0..31 from src tile, K 32..63 from dst tile
      v16h Asrc = load_arow(hb + (size_t)nsrc * 32, lane);
      v16h Adst = load_arow(hb + (size_t)ndst * 32, lane);

      v8f c0 = splat8(b1lo);
      v8f c1 = splat8(b1hi);
      c0 = wmma_f16_16x16x32(Asrc, B1[0][0], c0);
      c0 = wmma_f16_16x16x32(Adst, B1[1][0], c0);
      c1 = wmma_f16_16x16x32(Asrc, B1[0][1], c1);
      c1 = wmma_f16_16x16x32(Adst, B1[1][1], c1);

      // relu; C-layout -> packed-permuted row-major f16 tile in LDS:
      // dword n of each row holds (feat n, feat n+16) -> one b32 store.
#pragma unroll
      for (int j = 0; j < 8; ++j) {
        const int row = j + ((lane >> 4) << 3);
        v2h pk;
        pk[0] = (_Float16)fmaxf(c0[j], 0.0f);
        pk[1] = (_Float16)fmaxf(c1[j], 0.0f);
        *(v2h*)&myt[row * 32 + 2 * (lane & 15)] = pk;
      }
      asm volatile("s_wait_dscnt 0" ::: "memory");
      v16h Am = load_arow(myt + (size_t)(lane & 15) * 32, lane);

      v8f d0 = splat8(b2lo);
      v8f d1 = splat8(b2hi);
      d0 = wmma_f16_16x16x32(Am, B2[0], d0);
      d1 = wmma_f16_16x16x32(Am, B2[1], d1);

      // leaky + symmetric scatter-add (message goes to BOTH endpoints).
      float* sb = sums + (size_t)b * Nn * 32;
      const int f0 = lane & 15, f1 = 16 + (lane & 15);
#pragma unroll
      for (int j = 0; j < 8; ++j) {
        float m0 = d0[j]; m0 = m0 > 0.0f ? m0 : LEAKY_SLOPE * m0;
        float m1 = d1[j]; m1 = m1 > 0.0f ? m1 : LEAKY_SLOPE * m1;
        atomicAdd(&sb[(size_t)drow[j] * 32 + f0], m0);
        atomicAdd(&sb[(size_t)drow[j] * 32 + f1], m1);
        atomicAdd(&sb[(size_t)srow[j] * 32 + f0], m0);
        atomicAdd(&sb[(size_t)srow[j] * 32 + f1], m1);
      }
    }
  }
}

// Scalar fallback for E % 16 edges (not taken for E=160000; host-guarded).
__global__ void k_edge_rem(const _Float16* __restrict__ h16,
                           const int* __restrict__ e0, const int* __restrict__ e1,
                           const float* __restrict__ We1, const float* __restrict__ be1,
                           const float* __restrict__ We2, const float* __restrict__ be2,
                           float* __restrict__ sums, int Nn, int E, int E0, int Bb) {
  const int rem = E - E0;
  const int idx = blockIdx.x * blockDim.x + threadIdx.x;
  if (idx >= rem * Bb) return;
  const int e = E0 + idx % rem;
  const int b = idx / rem;
  const int s = e0[e], d = e1[e];
  const _Float16* hs = h16 + ((size_t)b * Nn + s) * 32;
  const _Float16* hd = h16 + ((size_t)b * Nn + d) * 32;
  float pair[64];
  for (int i = 0; i < 32; ++i) { pair[i] = (float)hs[i]; pair[32 + i] = (float)hd[i]; }
  float m1[32];
  for (int o = 0; o < 32; ++o) {
    float a = be1[o];
    for (int k = 0; k < 64; ++k) a = fmaf(We1[o * 64 + k], pair[k], a);
    m1[o] = fmaxf(a, 0.0f);
  }
  for (int o = 0; o < 32; ++o) {
    float a = be2[o];
    for (int k = 0; k < 32; ++k) a = fmaf(We2[o * 32 + k], m1[k], a);
    a = a > 0.0f ? a : LEAKY_SLOPE * a;
    atomicAdd(&sums[((size_t)b * Nn + d) * 32 + o], a);
    atomicAdd(&sums[((size_t)b * Nn + s) * 32 + o], a);
  }
}

// ---------------------------------------------------------------------------
// Stage 3: agg = where(cnt>0, sums/cnt, 0); MLP 32->32 (relu, WMMA) ->1
// (leaky, per-row reduce from LDS); accumulate v[b,n]*Wg[n] into gpart[b].
// The relu'd layer-1 tile is stored K-permuted as packed float2 (b64 stores);
// the 32->1 reduce indexes Wv2 through kperm() to compensate.
// ---------------------------------------------------------------------------
__global__ void __launch_bounds__(32 * WAVES_PER_BLOCK)
k_vertex_out(const float* __restrict__ sums, const float* __restrict__ cnt,
             const float* __restrict__ Wv1, const float* __restrict__ bv1,
             const float* __restrict__ Wv2, const float* __restrict__ bv2,
             const float* __restrict__ Wg, float* __restrict__ gpart, int Nn,
             int Bb, int ntiles) {
  __shared__ __align__(16) float ldsf[WAVES_PER_BLOCK * 16 * 32];
  const int lane = threadIdx.x & 31;
  const int wib  = threadIdx.x >> 5;
  float* myt = &ldsf[wib * 512];
  const int gw = blockIdx.x * WAVES_PER_BLOCK + wib;
  const int nw = gridDim.x * WAVES_PER_BLOCK;

  v16h Bv[2];
  Bv[0] = make_btile(Wv1, 32, 0, 0, lane);
  Bv[1] = make_btile(Wv1, 32, 16, 0, lane);
  const float bvlo = bv1[lane & 15], bvhi = bv1[16 + (lane & 15)];
  const float bv2s = bv2[0];

  // Wv2 pre-permuted to the packed tile order (registers, lane-uniform).
  float w2p[32];
#pragma unroll
  for (int p = 0; p < 32; ++p) w2p[p] = Wv2[kperm(p)];

  for (int t = gw; t < ntiles; t += nw) {
    const int nbase = t * 16;
    const int myn   = nbase + (lane & 15);
    const float c   = cnt[myn];
    const float scl = c > 0.0f ? 1.0f / c : 0.0f;  // where(cnt>0, /cnt, 0)

    for (int b = 0; b < Bb; ++b) {
      const float* sp =
          sums + ((size_t)b * Nn + myn) * 32 + ((lane >> 4) << 3);
      v16h A;
#pragma unroll
      for (int v = 0; v < 8; ++v) {
        A[v]     = (_Float16)(sp[v] * scl);
        A[8 + v] = (_Float16)(sp[v + 16] * scl);
      }
      v8f c0 = splat8(bvlo);
      v8f c1 = splat8(bvhi);
      c0 = wmma_f16_16x16x32(A, Bv[0], c0);
      c1 = wmma_f16_16x16x32(A, Bv[1], c1);
#pragma unroll
      for (int j = 0; j < 8; ++j) {
        const int row = j + ((lane >> 4) << 3);
        v2f pk;
        pk[0] = fmaxf(c0[j], 0.0f);
        pk[1] = fmaxf(c1[j], 0.0f);
        *(v2f*)&myt[row * 32 + 2 * (lane & 15)] = pk;  // ds_store_b64
      }
      asm volatile("s_wait_dscnt 0" ::: "memory");
      if (lane < 16) {
        float acc = bv2s;
#pragma unroll
        for (int p = 0; p < 32; ++p) acc = fmaf(w2p[p], myt[lane * 32 + p], acc);
        const float v = acc > 0.0f ? acc : LEAKY_SLOPE * acc;
        atomicAdd(&gpart[b], v * Wg[nbase + lane]);
      }
    }
  }
}

// Scalar fallback for N % 16 nodes (not taken for N=20000; host-guarded).
__global__ void k_vertex_rem(const float* __restrict__ sums,
                             const float* __restrict__ cnt,
                             const float* __restrict__ Wv1, const float* __restrict__ bv1,
                             const float* __restrict__ Wv2, const float* __restrict__ bv2,
                             const float* __restrict__ Wg, float* __restrict__ gpart,
                             int Nn, int N0, int Bb) {
  const int rem = Nn - N0;
  const int idx = blockIdx.x * blockDim.x + threadIdx.x;
  if (idx >= rem * Bb) return;
  const int n = N0 + idx % rem;
  const int b = idx / rem;
  const float c = cnt[n];
  const float scl = c > 0.0f ? 1.0f / c : 0.0f;
  const float* sp = sums + ((size_t)b * Nn + n) * 32;
  float h1[32];
  for (int o = 0; o < 32; ++o) {
    float a = bv1[o];
    for (int k = 0; k < 32; ++k) a = fmaf(Wv1[o * 32 + k], sp[k] * scl, a);
    h1[o] = fmaxf(a, 0.0f);
  }
  float acc = bv2[0];
  for (int f = 0; f < 32; ++f) acc = fmaf(Wv2[f], h1[f], acc);
  const float v = acc > 0.0f ? acc : LEAKY_SLOPE * acc;
  atomicAdd(&gpart[b], v * Wg[n]);
}

// ---------------------------------------------------------------------------
// Stage 4: sigmoid readout.
// ---------------------------------------------------------------------------
__global__ void k_sigmoid(const float* __restrict__ gpart,
                          const float* __restrict__ bg, float* __restrict__ out,
                          int Bb) {
  const int b = blockIdx.x * blockDim.x + threadIdx.x;
  if (b < Bb) out[b] = 1.0f / (1.0f + __expf(-(gpart[b] + bg[0])));
}

// ---------------------------------------------------------------------------
extern "C" void kernel_launch(void* const* d_in, const int* in_sizes, int n_in,
                              void* d_out, int out_size, void* d_ws,
                              size_t ws_size, hipStream_t stream) {
  const float* verts = (const float*)d_in[0];
  const int*   edges = (const int*)d_in[1];
  const float* Wx1 = (const float*)d_in[2];
  const float* bx1 = (const float*)d_in[3];
  const float* Wx2 = (const float*)d_in[4];
  const float* bx2 = (const float*)d_in[5];
  const float* We1 = (const float*)d_in[6];
  const float* be1 = (const float*)d_in[7];
  const float* We2 = (const float*)d_in[8];
  const float* be2 = (const float*)d_in[9];
  const float* Wv1 = (const float*)d_in[10];
  const float* bv1 = (const float*)d_in[11];
  const float* Wv2 = (const float*)d_in[12];
  const float* bv2 = (const float*)d_in[13];
  const float* Wg  = (const float*)d_in[14];
  const float* bg  = (const float*)d_in[15];

  const int Nn = in_sizes[14];           // Wg: (1, N)
  const int E  = in_sizes[1] / 2;        // edges: (2, E)
  const int Bb = in_sizes[0] / (Nn * 11);
  const int* e0 = edges;
  const int* e1 = edges + E;

  // Workspace layout (256B-aligned): h16 | sums | cnt | gpart
  char* ws = (char*)d_ws;
  const size_t h16_bytes  = (size_t)Bb * Nn * 32 * sizeof(_Float16);
  const size_t sums_bytes = (size_t)Bb * Nn * 32 * sizeof(float);
  size_t off = 0;
  _Float16* h16 = (_Float16*)(ws + off);
  off += (h16_bytes + 255) & ~(size_t)255;
  float* sums = (float*)(ws + off);
  off += (sums_bytes + 255) & ~(size_t)255;
  float* cnt = (float*)(ws + off);
  off += (((size_t)Nn * sizeof(float)) + 255) & ~(size_t)255;
  float* gpart = (float*)(ws + off);

  hipMemsetAsync(sums, 0, sums_bytes, stream);
  hipMemsetAsync(cnt, 0, (size_t)Nn * sizeof(float), stream);
  hipMemsetAsync(gpart, 0, (size_t)Bb * sizeof(float), stream);

  const int BN = Bb * Nn;
  k_node_mlp<<<(BN + 255) / 256, 256, 0, stream>>>(verts, Wx1, bx1, Wx2, bx2,
                                                   h16, BN);
  k_count<<<(E + 255) / 256, 256, 0, stream>>>(e0, e1, cnt, E);

  const int etiles = E / 16;
  if (etiles > 0)
    k_edge_mlp<<<512, 32 * WAVES_PER_BLOCK, 0, stream>>>(
        h16, e0, e1, We1, be1, We2, be2, sums, Nn, Bb, etiles);
  if (E % 16) {
    const int tot = (E % 16) * Bb;
    k_edge_rem<<<(tot + 127) / 128, 128, 0, stream>>>(
        h16, e0, e1, We1, be1, We2, be2, sums, Nn, E, etiles * 16, Bb);
  }

  const int vtiles = Nn / 16;
  if (vtiles > 0)
    k_vertex_out<<<256, 32 * WAVES_PER_BLOCK, 0, stream>>>(
        sums, cnt, Wv1, bv1, Wv2, bv2, Wg, gpart, Nn, Bb, vtiles);
  if (Nn % 16) {
    const int tot = (Nn % 16) * Bb;
    k_vertex_rem<<<(tot + 127) / 128, 128, 0, stream>>>(
        sums, cnt, Wv1, bv1, Wv2, bv2, Wg, gpart, Nn, vtiles * 16, Bb);
  }

  k_sigmoid<<<1, 64, 0, stream>>>(gpart, bg, (float*)d_out, Bb);
}
